// DGCNN_37065567765039
// MI455X (gfx1250) — matrix operand order repaired and checked by reading, not compiled, not run on hardware
//
#include <hip/hip_runtime.h>
#include <hip/hip_bf16.h>

typedef __attribute__((ext_vector_type(16))) _Float16 v16h;
typedef __attribute__((ext_vector_type(8)))  _Float16 v8h;
typedef __attribute__((ext_vector_type(8)))  float    v8f;

#define BCLOUDS 32
#define NPTS    1024
#define KNN     30
#define NROWS   (BCLOUDS * NPTS)   // 32768
#define FEATW   320

// ---------------------------------------------------------------------------
// WMMA tile helpers (CDNA5 wave32 layouts, ISA 7.12.2).
// A 16x32 f16 from row-major f16 (lda in halfs): per lane two contiguous
// 8-half groups -> two b128 loads, no branches.
// ---------------------------------------------------------------------------
__device__ __forceinline__ v16h load_a16(const _Float16* __restrict__ X, int lda,
                                         int mBase, int kBase) {
    int lane = threadIdx.x & 31;
    const _Float16* p = X + (long)(mBase + (lane & 15)) * lda + kBase + ((lane >> 4) << 4);
    v8h lo = *(const v8h*)p;
    v8h hi = *(const v8h*)(p + 8);
    v16h r;
#pragma unroll
    for (int i = 0; i < 8; ++i) { r[i] = lo[i]; r[8 + i] = hi[i]; }
    return r;
}

// B pre-packed: per (kTile,nTile) block of 32x16, each lane's 16 halfs contiguous.
__device__ __forceinline__ v16h load_b_packed(const _Float16* __restrict__ Bpk, int blk) {
    int lane = threadIdx.x & 31;
    const _Float16* p = Bpk + (((long)blk * 32) + lane) * 16;
    v8h lo = *(const v8h*)p;
    v8h hi = *(const v8h*)(p + 8);
    v16h r;
#pragma unroll
    for (int i = 0; i < 8; ++i) { r[i] = lo[i]; r[8 + i] = hi[i]; }
    return r;
}

// D 16x16 f32: lane n=(lane&15) is column; VGPR r is row r (lanes<16) or 8+r.
__device__ __forceinline__ void store_d(float* __restrict__ Out, int ldo,
                                        int mBase, int nBase, v8f c) {
    int lane = threadIdx.x & 31;
    int n    = nBase + (lane & 15);
    int mOff = (lane >> 4) * 8;
#pragma unroll
    for (int r = 0; r < 8; ++r)
        Out[(long)(mBase + mOff + r) * ldo + n] = c[r];
}

// ---------------------------------------------------------------------------
// kNN: one wave32 per point; 32 candidate distances per lane in VGPRs,
// 30 rounds of local scan + wave-min butterfly.
// ---------------------------------------------------------------------------
__global__ void knn_kernel(const float* __restrict__ X, int lda, int d,
                           int* __restrict__ Idx) {
    int row  = blockIdx.x;
    int b    = row >> 10;
    int lane = threadIdx.x;
    const float* xi = X + (long)row * lda;            // block-uniform -> scalar loads
    const float* Xb = X + ((long)b << 10) * lda;

    float dist[32];
#pragma unroll
    for (int t = 0; t < 32; ++t) {
        const float* xj = Xb + (long)(t * 32 + lane) * lda;
        float acc = 0.0f;
        for (int dd = 0; dd < d; ++dd) {
            float df = xi[dd] - xj[dd];
            acc += df * df;
        }
        dist[t] = acc;
    }

    for (int kk = 0; kk < KNN; ++kk) {
        float best = 1e30f; int bestT = 0;
#pragma unroll
        for (int t = 0; t < 32; ++t)
            if (dist[t] < best) { best = dist[t]; bestT = t; }
        int bestJ = bestT * 32 + lane;
#pragma unroll
        for (int s = 16; s > 0; s >>= 1) {
            float ov = __shfl_xor(best, s, 32);
            int   oj = __shfl_xor(bestJ, s, 32);
            if (ov < best || (ov == best && oj < bestJ)) { best = ov; bestJ = oj; }
        }
        if (lane == 0) Idx[(long)row * KNN + kk] = bestJ;
        if ((bestJ & 31) == lane) {
            int t = bestJ >> 5;
#pragma unroll
            for (int u = 0; u < 32; ++u)
                dist[u] = (u == t) ? 1e30f : dist[u];
        }
    }
}

// ---------------------------------------------------------------------------
// EdgeConv GEMM: UV[32768 x Nt] = X[32768 x 32*KT] @ B  (f16 WMMA, guard-free)
// 8 waves/block; each wave: KT B-tiles resident, 4 M-tiles as 2 interleaved
// pairs (independent accumulators -> loads of one tile overlap WMMA of other).
// ---------------------------------------------------------------------------
template<int KT>
__global__ void edge_gemm(const _Float16* __restrict__ X, int lda,
                          const _Float16* __restrict__ Bpk, int nTiles,
                          float* __restrict__ UV, int Nt) {
    int wave   = threadIdx.x >> 5;
    int nT     = blockIdx.y;
    int mBase0 = (blockIdx.x * 8 + wave) * 64;

    v16h btile[KT];
#pragma unroll
    for (int kt = 0; kt < KT; ++kt)
        btile[kt] = load_b_packed(Bpk, kt * nTiles + nT);

#pragma unroll
    for (int mp = 0; mp < 2; ++mp) {
        int mA = mBase0 + mp * 32;
        int mB = mA + 16;
        v8f c0 = {}, c1 = {};
#pragma unroll
        for (int kt = 0; kt < KT; ++kt) {
            v16h a0 = load_a16(X, lda, mA, kt * 32);
            v16h a1 = load_a16(X, lda, mB, kt * 32);
            c0 = __builtin_amdgcn_wmma_f32_16x16x32_f16(false, a0, false, btile[kt],
                                                        (short)0, c0, false, false);
            c1 = __builtin_amdgcn_wmma_f32_16x16x32_f16(false, a1, false, btile[kt],
                                                        (short)0, c1, false, false);
        }
        store_d(UV, Nt, mA, nT * 16, c0);
        store_d(UV, Nt, mB, nT * 16, c1);
    }
}

// out[row,c] = u[row,c] + bias[c] + max_{kk} v[nbr(row,kk), c];  f32 + f16 mirror
__global__ void edge_aggregate(const float* __restrict__ UV, int Nt,
                               const int* __restrict__ Idx,
                               const float* __restrict__ bias,
                               float* __restrict__ OutF,
                               _Float16* __restrict__ OutH,
                               int ldo, int Dout) {
    int row = blockIdx.x;
    int cb  = (row >> 10) << 10;
    int c   = threadIdx.x;
    float m = -1e30f;
    for (int kk = 0; kk < KNN; ++kk) {
        int j = Idx[(long)row * KNN + kk];            // block-uniform -> scalar load
        m = fmaxf(m, UV[(long)(cb + j) * Nt + Dout + c]);
    }
    float val = UV[(long)row * Nt + c] + bias[c] + m;
    OutF[(long)row * ldo + c] = val;
    OutH[(long)row * ldo + c] = (_Float16)val;
}

// ---------------------------------------------------------------------------
// lin1 + global max pool fused. grid (B, 64 n-tiles); 8 waves; 10 B-tiles
// (K=320) resident in regs; each wave covers 8 of 64 M-tiles, processed as
// interleaved pairs for load/WMMA overlap.
// ---------------------------------------------------------------------------
__global__ void lin1_pool(const _Float16* __restrict__ FeatH,
                          const _Float16* __restrict__ WlPk,
                          const float* __restrict__ bl,
                          float* __restrict__ Pooled) {
    __shared__ float red[8 * 16];
    int b    = blockIdx.x;
    int nT   = blockIdx.y;
    int wave = threadIdx.x >> 5;
    int lane = threadIdx.x & 31;

    v16h btile[10];
#pragma unroll
    for (int kt = 0; kt < 10; ++kt)
        btile[kt] = load_b_packed(WlPk, kt * 64 + nT);

    float runmax = -1e30f;
    for (int mt = wave; mt < 64; mt += 16) {          // pairs (mt, mt+8)
        int mA = (b << 10) + mt * 16;
        int mB = mA + 8 * 16;
        v8f c0 = {}, c1 = {};
#pragma unroll
        for (int kt = 0; kt < 10; ++kt) {
            v16h a0 = load_a16(FeatH, FEATW, mA, kt * 32);
            v16h a1 = load_a16(FeatH, FEATW, mB, kt * 32);
            c0 = __builtin_amdgcn_wmma_f32_16x16x32_f16(false, a0, false, btile[kt],
                                                        (short)0, c0, false, false);
            c1 = __builtin_amdgcn_wmma_f32_16x16x32_f16(false, a1, false, btile[kt],
                                                        (short)0, c1, false, false);
        }
        float t0 = c0[0], t1 = c1[0];
#pragma unroll
        for (int r = 1; r < 8; ++r) { t0 = fmaxf(t0, c0[r]); t1 = fmaxf(t1, c1[r]); }
        runmax = fmaxf(runmax, fmaxf(t0, t1));
    }
    runmax = fmaxf(runmax, __shfl_xor(runmax, 16, 32));   // combine row halves
    if (lane < 16) red[wave * 16 + lane] = runmax;
    __syncthreads();
    if (threadIdx.x < 16) {
        float m = red[threadIdx.x];
        for (int w = 1; w < 8; ++w) m = fmaxf(m, red[w * 16 + threadIdx.x]);
        Pooled[b * 1024 + nT * 16 + threadIdx.x] = m + bl[nT * 16 + threadIdx.x];
    }
}

// ---------------------------------------------------------------------------
// MLP head (32 rows) — f32 VALU, one thread per output element.
// ---------------------------------------------------------------------------
__global__ void head_layer(const float* __restrict__ In, int M, int Kd,
                           const float* __restrict__ W, const float* __restrict__ bias,
                           const float* __restrict__ g, const float* __restrict__ be,
                           float* __restrict__ Out, int Nd, int relu_bn) {
    int i = blockIdx.x * blockDim.x + threadIdx.x;
    if (i >= M * Nd) return;
    int m = i / Nd, n = i % Nd;
    const float* in = In + (long)m * Kd;
    float acc = 0.0f;
    for (int k = 0; k < Kd; ++k) acc += in[k] * W[(long)k * Nd + n];
    acc += bias[n];
    if (relu_bn) {
        const float inv = 0.99999500003749969f;   // 1/sqrt(1+1e-5)
        acc = g[n] * acc * inv + be[n];
        acc = fmaxf(acc, 0.0f);
    }
    Out[i] = acc;
}

// ---------------------------------------------------------------------------
// Prep: EdgeConv combined weight  Wc[k,c] = Wtop-Wbot | Wbot, K zero-padded to
// multiple of 32, scattered directly into the packed WMMA-B layout.
// ---------------------------------------------------------------------------
__global__ void build_wc_packed(const float* __restrict__ W, int Din, int Dout,
                                _Float16* __restrict__ Bpk) {
    int N    = 2 * Dout;
    int Kpad = (Din + 31) & ~31;
    int i = blockIdx.x * blockDim.x + threadIdx.x;
    if (i >= Kpad * N) return;
    int k = i / N, c = i % N;
    float val = 0.0f;
    if (k < Din)
        val = (c < Dout) ? (W[(long)k * Dout + c] - W[(long)(Din + k) * Dout + c])
                         : W[(long)(Din + k) * Dout + (c - Dout)];
    int kT = k >> 5, nT = c >> 4, kr = k & 31, nc = c & 15;
    int nTiles = N >> 4;
    Bpk[(((long)(kT * nTiles + nT) * 32) + nc + (kr & 16)) * 16 + (kr & 15)] = (_Float16)val;
}

// Pack a plain row-major f32 [K x N] weight into the packed WMMA-B f16 layout.
__global__ void pack_b(const float* __restrict__ W, int K, int N,
                       _Float16* __restrict__ Bpk) {
    int i = blockIdx.x * blockDim.x + threadIdx.x;
    if (i >= K * N) return;
    int k = i / N, n = i % N;
    int kT = k >> 5, nT = n >> 4, kr = k & 31, nc = n & 15;
    int nTiles = N >> 4;
    Bpk[(((long)(kT * nTiles + nT) * 32) + nc + (kr & 16)) * 16 + (kr & 15)] = (_Float16)W[i];
}

// pos [32768x3] f32 -> posH [32768x32] f16, zero-padded K (for guard-free WMMA)
__global__ void pos_to_h(const float* __restrict__ pos, _Float16* __restrict__ posH) {
    int i = blockIdx.x * blockDim.x + threadIdx.x;
    if (i >= NROWS * 32) return;
    int r = i >> 5, c = i & 31;
    posH[i] = (_Float16)((c < 3) ? pos[r * 3 + c] : 0.0f);
}

// ---------------------------------------------------------------------------
extern "C" void kernel_launch(void* const* d_in, const int* in_sizes, int n_in,
                              void* d_out, int out_size, void* d_ws, size_t ws_size,
                              hipStream_t stream) {
    const float* pos = (const float*)d_in[0];
    const float* W1 = (const float*)d_in[2];  const float* b1 = (const float*)d_in[3];
    const float* W2 = (const float*)d_in[4];  const float* b2 = (const float*)d_in[5];
    const float* W3 = (const float*)d_in[6];  const float* b3 = (const float*)d_in[7];
    const float* W4 = (const float*)d_in[8];  const float* b4 = (const float*)d_in[9];
    const float* Wl = (const float*)d_in[10]; const float* bl = (const float*)d_in[11];
    const float* Wm1 = (const float*)d_in[12]; const float* bm1 = (const float*)d_in[13];
    const float* g1 = (const float*)d_in[14];  const float* be1 = (const float*)d_in[15];
    const float* Wm2 = (const float*)d_in[16]; const float* bm2 = (const float*)d_in[17];
    const float* g2 = (const float*)d_in[18];  const float* be2 = (const float*)d_in[19];
    const float* Wm3 = (const float*)d_in[20]; const float* bm3 = (const float*)d_in[21];
    float* out = (float*)d_out;

    char* ws = (char*)d_ws;
    size_t off = 0;
    float*    feat  = (float*)(ws + off);    off += (size_t)NROWS * FEATW * 4;  // 41.9 MB
    _Float16* featH = (_Float16*)(ws + off); off += (size_t)NROWS * FEATW * 2;  // 21.0 MB
    float*    uv    = (float*)(ws + off);    off += (size_t)NROWS * 256 * 4;    // 33.6 MB
    int*      idx   = (int*)(ws + off);      off += (size_t)NROWS * KNN * 4;    //  3.9 MB
    _Float16* posH  = (_Float16*)(ws + off); off += (size_t)NROWS * 32 * 2;     //  2.1 MB
    _Float16* WlPk  = (_Float16*)(ws + off); off += (size_t)FEATW * 1024 * 2;
    _Float16* WcPk  = (_Float16*)(ws + off); off += (size_t)64 * 256 * 2;
    float* pooled = (float*)(ws + off);      off += (size_t)32 * 1024 * 4;
    float* h1     = (float*)(ws + off);      off += (size_t)32 * 512 * 4;
    float* h2     = (float*)(ws + off);      off += (size_t)32 * 256 * 4;

    // ---- EdgeConv 1: pos -> feat[:,0:64]   (K padded 3->32)
    pos_to_h<<<dim3((NROWS * 32 + 255) / 256), 256, 0, stream>>>(pos, posH);
    build_wc_packed<<<dim3((32 * 128 + 255) / 256), 256, 0, stream>>>(W1, 3, 64, WcPk);
    knn_kernel<<<dim3(NROWS), 32, 0, stream>>>(pos, 3, 3, idx);
    edge_gemm<1><<<dim3(64, 8), 256, 0, stream>>>(posH, 32, WcPk, 8, uv, 128);
    edge_aggregate<<<dim3(NROWS), 64, 0, stream>>>(uv, 128, idx, b1,
                                                   feat + 0, featH + 0, FEATW, 64);

    // ---- EdgeConv 2: feat[:,0:64] -> feat[:,64:128]
    build_wc_packed<<<dim3((64 * 128 + 255) / 256), 256, 0, stream>>>(W2, 64, 64, WcPk);
    knn_kernel<<<dim3(NROWS), 32, 0, stream>>>(feat + 0, FEATW, 64, idx);
    edge_gemm<2><<<dim3(64, 8), 256, 0, stream>>>(featH + 0, FEATW, WcPk, 8, uv, 128);
    edge_aggregate<<<dim3(NROWS), 64, 0, stream>>>(uv, 128, idx, b2,
                                                   feat + 64, featH + 64, FEATW, 64);

    // ---- EdgeConv 3: feat[:,64:128] -> feat[:,128:192]
    build_wc_packed<<<dim3((64 * 128 + 255) / 256), 256, 0, stream>>>(W3, 64, 64, WcPk);
    knn_kernel<<<dim3(NROWS), 32, 0, stream>>>(feat + 64, FEATW, 64, idx);
    edge_gemm<2><<<dim3(64, 8), 256, 0, stream>>>(featH + 64, FEATW, WcPk, 8, uv, 128);
    edge_aggregate<<<dim3(NROWS), 64, 0, stream>>>(uv, 128, idx, b3,
                                                   feat + 128, featH + 128, FEATW, 64);

    // ---- EdgeConv 4: feat[:,128:192] -> feat[:,192:320]
    build_wc_packed<<<dim3((64 * 256 + 255) / 256), 256, 0, stream>>>(W4, 64, 128, WcPk);
    knn_kernel<<<dim3(NROWS), 32, 0, stream>>>(feat + 128, FEATW, 64, idx);
    edge_gemm<2><<<dim3(64, 16), 256, 0, stream>>>(featH + 128, FEATW, WcPk, 16, uv, 256);
    edge_aggregate<<<dim3(NROWS), 128, 0, stream>>>(uv, 256, idx, b4,
                                                    feat + 192, featH + 192, FEATW, 128);

    // ---- lin1 (320->1024) fused with global max pool
    pack_b<<<dim3((FEATW * 1024 + 255) / 256), 256, 0, stream>>>(Wl, FEATW, 1024, WlPk);
    lin1_pool<<<dim3(32, 64), 256, 0, stream>>>(featH, WlPk, bl, pooled);

    // ---- MLP head
    head_layer<<<dim3((32 * 512 + 255) / 256), 256, 0, stream>>>(
        pooled, 32, 1024, Wm1, bm1, g1, be1, h1, 512, 1);
    head_layer<<<dim3((32 * 256 + 255) / 256), 256, 0, stream>>>(
        h1, 32, 512, Wm2, bm2, g2, be2, h2, 256, 1);
    head_layer<<<dim3((32 * 40 + 255) / 256), 256, 0, stream>>>(
        h2, 32, 256, Wm3, bm3, nullptr, nullptr, out, 40, 0);
}
